// EdgeSamplingHead_35218731827670
// MI455X (gfx1250) — matrix operand order repaired and checked by reading, not compiled, not run on hardware
//
#include <hip/hip_runtime.h>
#include <hip/hip_bf16.h>
#include <stdint.h>

#define N_NODES 8192
#define E_EDGES 262144
#define D_FEAT  128
#define HID_MLP 256
#define K_KEEP  131072

typedef __bf16   bf16x16 __attribute__((ext_vector_type(16)));
typedef float    f32x8   __attribute__((ext_vector_type(8)));
typedef float    f32x4v  __attribute__((ext_vector_type(4)));
typedef unsigned int u32x4v __attribute__((ext_vector_type(4)));
typedef int      i32x8v  __attribute__((ext_vector_type(8)));
typedef int      i32x4v  __attribute__((ext_vector_type(4)));

union Frag { uint32_t u[8]; bf16x16 b; };

__device__ __forceinline__ uint32_t f2bf_rne(float f) {
    uint32_t u = __float_as_uint(f);
    return (u + 0x7FFFu + ((u >> 16) & 1u)) >> 16;
}
__device__ __forceinline__ uint32_t pack2bf(float lo, float hi) {
    return f2bf_rne(lo) | (f2bf_rne(hi) << 16);
}
__device__ __forceinline__ float bf2f(uint32_t b16) { return __uint_as_float(b16 << 16); }
__device__ __forceinline__ uint32_t addbf_pair(uint32_t a, uint32_t b) {
    float lo = bf2f(a & 0xffffu) + bf2f(b & 0xffffu);
    float hi = bf2f(a >> 16) + bf2f(b >> 16);
    return pack2bf(lo, hi);
}
__device__ __forceinline__ f32x8 wmma_bf16(const Frag& a, const Frag& b, f32x8 c) {
    return __builtin_amdgcn_wmma_f32_16x16x32_bf16(false, a.b, false, b.b, (short)0, c, false, false);
}
__device__ __forceinline__ void load_bfrag(Frag& f, const uint32_t* base) {
    const uint4* p = (const uint4*)base;
    uint4 q0 = p[0], q1 = p[1];
    f.u[0] = q0.x; f.u[1] = q0.y; f.u[2] = q0.z; f.u[3] = q0.w;
    f.u[4] = q1.x; f.u[5] = q1.y; f.u[6] = q1.z; f.u[7] = q1.w;
}

// Issue one TDM transfer: 2D tile, 32 fp32 cols x 128 rows, row stride 8192 fp32.
__device__ __forceinline__ void tdm_issue(const float* gsrc, uint32_t ldsAddr) {
    uint64_t ga = (uint64_t)(uintptr_t)gsrc;
    u32x4v g0; i32x8v g1; i32x4v g2; i32x4v g3;
    g0[0] = 1u;                                             // count=1 user descriptor
    g0[1] = ldsAddr;                                        // lds_addr
    g0[2] = (uint32_t)ga;                                   // global_addr[31:0]
    g0[3] = (uint32_t)((ga >> 32) & 0x01FFFFFFu) | (2u << 30); // addr hi | type=2
    g1[0] = (int)(2u << 16);                                // data_size=4B
    g1[1] = (int)(8192u << 16);                             // tensor_dim0 lo16
    g1[2] = (int)(8192u << 16);                             // dim0 hi | tensor_dim1 lo16
    g1[3] = (int)(32u << 16);                               // dim1 hi | tile_dim0=32
    g1[4] = 128;                                            // tile_dim1=128, tile_dim2=0
    g1[5] = 8192;                                           // tensor_dim0_stride lo32
    g1[6] = 0; g1[7] = 0;
    g2[0] = g2[1] = g2[2] = g2[3] = 0;
    g3[0] = g3[1] = g3[2] = g3[3] = 0;
#if defined(__clang_major__) && (__clang_major__ >= 23)
    i32x8v g4; for (int q = 0; q < 8; ++q) g4[q] = 0;
    __builtin_amdgcn_tensor_load_to_lds(g0, g1, g2, g3, g4, 0);
#else
    __builtin_amdgcn_tensor_load_to_lds(g0, g1, g2, g3, 0);
#endif
}

// ---------------- zero helpers ----------------
__global__ void k_zero_u32(uint32_t* __restrict__ p, int n) {
    int i = blockIdx.x * 256 + threadIdx.x;
    if (i < n) p[i] = 0u;
}
__global__ void k_zero_out(f32x4v* __restrict__ p) {
    int i = blockIdx.x * 256 + threadIdx.x;   // 65536*256 vec4 = 256 MB
    f32x4v z = {0.f, 0.f, 0.f, 0.f};
    __builtin_nontemporal_store(z, p + i);    // don't pollute L2 (A is re-read later)
}

// ---------------- swizzle a K x N fp32 matrix into WMMA bf16 B-fragment layout ----------------
// frag index: (((kt*NT + nt)*32 + lane)*8 + j); lane: h=lane>>4, n=(lane&15)+16*nt
// dword j of lane holds B[kt*32 + 8h + 2j + 8*(j>>2), n], B[kt*32 + 8h + 2j + 8*(j>>2) + 1, n] packed bf16
__global__ void k_swizzle_b(const float* __restrict__ B, uint32_t* __restrict__ frag,
                            int K, int N, int NT, int total) {
    int id = blockIdx.x * 256 + threadIdx.x;
    if (id >= total) return;
    int j = id & 7;
    int lane = (id >> 3) & 31;
    int rest = id >> 8;
    int nt = rest % NT;
    int kt = rest / NT;
    int h = lane >> 4;
    int n = (lane & 15) + nt * 16;
    int k0 = kt * 32 + 8 * h + 2 * j + 8 * (j >> 2);
    float v0 = (k0     < K) ? B[(size_t)k0 * N + n]       : 0.f;
    float v1 = (k0 + 1 < K) ? B[(size_t)(k0 + 1) * N + n] : 0.f;
    frag[id] = pack2bf(v0, v1);
}

// ---------------- K1: XW = X @ W_gnn  (fp32 out, 8192x128) ----------------
__global__ __launch_bounds__(256) void k_gemm_xw(const float* __restrict__ X,
                                                 const uint32_t* __restrict__ wgfrag,
                                                 float* __restrict__ XW) {
    const int lane = threadIdx.x & 31, w = threadIdx.x >> 5;
    const int h = lane >> 4, m = lane & 15;
    const int rowA = blockIdx.x * 128 + w * 16 + m;
    f32x8 acc[8];
    for (int i = 0; i < 8; ++i) for (int r = 0; r < 8; ++r) acc[i][r] = 0.f;
    for (int kt = 0; kt < 4; ++kt) {
        // batch-load all 8 B fragments first (one clause), then A, then 8 WMMAs
        Frag bfr[8];
        #pragma unroll
        for (int nt = 0; nt < 8; ++nt)
            load_bfrag(bfr[nt], wgfrag + (size_t)(((kt * 8 + nt) * 32 + lane) * 8));
        Frag a;
        const float* pa = X + (size_t)rowA * 128 + kt * 32 + 8 * h;
        float4 f0 = *(const float4*)(pa);
        float4 f1 = *(const float4*)(pa + 4);
        float4 f2 = *(const float4*)(pa + 16);
        float4 f3 = *(const float4*)(pa + 20);
        a.u[0] = pack2bf(f0.x, f0.y); a.u[1] = pack2bf(f0.z, f0.w);
        a.u[2] = pack2bf(f1.x, f1.y); a.u[3] = pack2bf(f1.z, f1.w);
        a.u[4] = pack2bf(f2.x, f2.y); a.u[5] = pack2bf(f2.z, f2.w);
        a.u[6] = pack2bf(f3.x, f3.y); a.u[7] = pack2bf(f3.z, f3.w);
        #pragma unroll
        for (int nt = 0; nt < 8; ++nt) acc[nt] = wmma_bf16(a, bfr[nt], acc[nt]);
    }
    for (int nt = 0; nt < 8; ++nt) {
        int n = m + nt * 16;
        #pragma unroll
        for (int r = 0; r < 8; ++r) {
            int rg = blockIdx.x * 128 + w * 16 + r + 8 * h;
            XW[(size_t)rg * 128 + n] = acc[nt][r];
        }
    }
}

// ---------------- K2: H = relu(A @ XW + b_gnn), bf16 out ----------------
// A tiles streamed by TDM with double buffering (issue kt+1 while computing kt).
__global__ __launch_bounds__(256) void k_gcn(const float* __restrict__ A,
                                              const uint32_t* __restrict__ xwfrag,
                                              const float* __restrict__ bg,
                                              uint16_t* __restrict__ Hb) {
    __shared__ float Atile[2][128 * 32];   // 2 x 16 KB fp32 tiles
    const int lane = threadIdx.x & 31, w = threadIdx.x >> 5;
    const int h = lane >> 4, m = lane & 15;
    const int row0 = blockIdx.x * 128;
    f32x8 acc[8];
    for (int i = 0; i < 8; ++i) for (int r = 0; r < 8; ++r) acc[i][r] = 0.f;
    const uint32_t lds0 = (uint32_t)(uintptr_t)(&Atile[0][0]);
    const uint32_t lds1 = (uint32_t)(uintptr_t)(&Atile[1][0]);
    const float* Arow = A + (size_t)row0 * 8192;

    if (threadIdx.x == 0) tdm_issue(Arow, lds0);            // prologue: tile 0 -> buf 0

    for (int kt = 0; kt < 256; ++kt) {
        if (threadIdx.x == 0) {
            if (kt + 1 < 256) {
                // overlap: DMA tile kt+1 into the other buffer, wait only for tile kt
                tdm_issue(Arow + (kt + 1) * 32, (kt & 1) ? lds0 : lds1);
                __builtin_amdgcn_s_wait_tensorcnt(1);
            } else {
                __builtin_amdgcn_s_wait_tensorcnt(0);
            }
        }
        __syncthreads();                                    // tile kt visible to all waves

        Frag bfr[8];
        #pragma unroll
        for (int nt = 0; nt < 8; ++nt)
            load_bfrag(bfr[nt], xwfrag + (size_t)(((kt * 8 + nt) * 32 + lane) * 8));

        Frag a;
        const float* pa = &Atile[kt & 1][(w * 16 + m) * 32 + 8 * h];
        float4 f0 = *(const float4*)(pa);
        float4 f1 = *(const float4*)(pa + 4);
        float4 f2 = *(const float4*)(pa + 16);
        float4 f3 = *(const float4*)(pa + 20);
        a.u[0] = pack2bf(f0.x, f0.y); a.u[1] = pack2bf(f0.z, f0.w);
        a.u[2] = pack2bf(f1.x, f1.y); a.u[3] = pack2bf(f1.z, f1.w);
        a.u[4] = pack2bf(f2.x, f2.y); a.u[5] = pack2bf(f2.z, f2.w);
        a.u[6] = pack2bf(f3.x, f3.y); a.u[7] = pack2bf(f3.z, f3.w);

        #pragma unroll
        for (int nt = 0; nt < 8; ++nt) acc[nt] = wmma_bf16(a, bfr[nt], acc[nt]);

        __syncthreads();   // all waves done reading buf kt&1 before it is re-filled at kt+2
    }
    for (int nt = 0; nt < 8; ++nt) {
        int n = m + nt * 16;
        float bn = bg[n];
        #pragma unroll
        for (int r = 0; r < 8; ++r) {
            int rg = row0 + w * 16 + r + 8 * h;
            float v = fmaxf(acc[nt][r] + bn, 0.f);
            Hb[(size_t)rg * 128 + n] = (uint16_t)f2bf_rne(v);
        }
    }
}

// ---------------- column mean of H -> h_G (bf16) ----------------
__global__ __launch_bounds__(256) void k_colmean(const uint16_t* __restrict__ Hb,
                                                 uint16_t* __restrict__ hGb) {
    __shared__ float red[256];
    int d = blockIdx.x, t = threadIdx.x;
    float s = 0.f;
    for (int r = t; r < N_NODES; r += 256) s += bf2f(Hb[(size_t)r * 128 + d]);
    red[t] = s; __syncthreads();
    for (int off = 128; off > 0; off >>= 1) {
        if (t < off) red[t] += red[t + off];
        __syncthreads();
    }
    if (t == 0) hGb[d] = (uint16_t)f2bf_rne(red[0] / (float)N_NODES);
}

// ---------------- K3: edge MLP + gumbel -> sortable keys ----------------
// block: 512 threads = 16 waves; 16 edges per block; wave w owns hidden ntile w
__global__ __launch_bounds__(512) void k_mlp(const uint16_t* __restrict__ Hb,
                                              const uint16_t* __restrict__ hGb,
                                              const uint32_t* __restrict__ w1frag,
                                              const float* __restrict__ b1,
                                              const float* __restrict__ W2,
                                              const float* __restrict__ b2,
                                              const int* __restrict__ ei,
                                              const int* __restrict__ ej,
                                              const float* __restrict__ noise,
                                              uint32_t* __restrict__ keys) {
    __shared__ float ypart[16];
    const int lane = threadIdx.x & 31, w = threadIdx.x >> 5;
    const int h = lane >> 4, m = lane & 15;
    if (threadIdx.x < 16) ypart[threadIdx.x] = 0.f;
    __syncthreads();
    const int et = blockIdx.x;
    const int e = et * 16 + m;
    const int ia = ei[e], ja = ej[e];
    const uint16_t* ri = Hb + (size_t)ia * 128;
    const uint16_t* rj = Hb + (size_t)ja * 128;
    f32x8 acc; for (int r = 0; r < 8; ++r) acc[r] = 0.f;

    #pragma unroll
    for (int kt = 0; kt < 9; ++kt) {
        Frag b;                                     // B load first: overlaps A assembly
        load_bfrag(b, w1frag + (size_t)(((kt * 16 + w) * 32 + lane) * 8));
        Frag a;
        if (kt < 4) {                               // H_v[i] + H_v[j] features
            int c0 = kt * 32 + 8 * h;
            uint4 pi0 = *(const uint4*)(ri + c0);
            uint4 pj0 = *(const uint4*)(rj + c0);
            uint4 pi1 = *(const uint4*)(ri + c0 + 16);
            uint4 pj1 = *(const uint4*)(rj + c0 + 16);
            a.u[0] = addbf_pair(pi0.x, pj0.x); a.u[1] = addbf_pair(pi0.y, pj0.y);
            a.u[2] = addbf_pair(pi0.z, pj0.z); a.u[3] = addbf_pair(pi0.w, pj0.w);
            a.u[4] = addbf_pair(pi1.x, pj1.x); a.u[5] = addbf_pair(pi1.y, pj1.y);
            a.u[6] = addbf_pair(pi1.z, pj1.z); a.u[7] = addbf_pair(pi1.w, pj1.w);
        } else if (kt < 8) {                        // broadcast h_G features
            int c0 = (kt - 4) * 32 + 8 * h;
            uint4 q0 = *(const uint4*)(hGb + c0);
            uint4 q1 = *(const uint4*)(hGb + c0 + 16);
            a.u[0] = q0.x; a.u[1] = q0.y; a.u[2] = q0.z; a.u[3] = q0.w;
            a.u[4] = q1.x; a.u[5] = q1.y; a.u[6] = q1.z; a.u[7] = q1.w;
        } else {                                    // K=256: constant 1.0, rest zero-pad
            for (int q = 0; q < 8; ++q) a.u[q] = 0u;
            if (h == 0) a.u[0] = 0x00003F80u;       // bf16(1.0) at K=256
        }
        acc = wmma_bf16(a, b, acc);
    }
    const int n = m + 16 * w;
    const float b1n = b1[n], w2n = W2[n];
    #pragma unroll
    for (int r = 0; r < 8; ++r) {
        float v = fmaxf(acc[r] + b1n, 0.f);        // relu layer 1
        atomicAdd(&ypart[r + 8 * h], v * w2n);     // fused layer-2 dot
    }
    __syncthreads();
    if (threadIdx.x < 16) {
        int ee = et * 16 + (int)threadIdx.x;
        float y = ypart[threadIdx.x] + b2[0];
        float u = noise[ee];
        float g = -logf(-logf(u + 1e-8f) + 1e-8f);
        uint32_t bits = __float_as_uint(y + g);
        keys[ee] = (bits & 0x80000000u) ? ~bits : (bits | 0x80000000u);
    }
}

// ---------------- radix-select top-K threshold ----------------
__global__ void k_hist_hi(const uint32_t* __restrict__ keys, uint32_t* __restrict__ hist) {
    int e = blockIdx.x * 256 + threadIdx.x;
    if (e < E_EDGES) atomicAdd(&hist[keys[e] >> 16], 1u);
}
__global__ __launch_bounds__(256) void k_scan_hi(const uint32_t* __restrict__ hist,
                                                 uint32_t* __restrict__ sel) {
    __shared__ uint32_t csum[256];
    int t = threadIdx.x;
    uint32_t s = 0;
    for (int b = 0; b < 256; ++b) s += hist[t * 256 + b];
    csum[t] = s; __syncthreads();
    if (t == 0) {
        uint32_t K = K_KEEP, acc = 0, b0 = 0, kp = 0;
        for (int c = 255; c >= 0; --c) {
            if (acc + csum[c] >= K) {
                for (int b = c * 256 + 255; b >= c * 256; --b) {
                    uint32_t hc = hist[b];
                    if (acc + hc >= K) { b0 = (uint32_t)b; kp = K - acc; break; }
                    acc += hc;
                }
                break;
            }
            acc += csum[c];
        }
        sel[0] = b0; sel[1] = kp;
    }
}
__global__ void k_hist_lo(const uint32_t* __restrict__ keys, const uint32_t* __restrict__ sel,
                          uint32_t* __restrict__ hist) {
    int e = blockIdx.x * 256 + threadIdx.x;
    if (e < E_EDGES) {
        uint32_t k = keys[e];
        if ((k >> 16) == sel[0]) atomicAdd(&hist[k & 0xffffu], 1u);
    }
}
__global__ __launch_bounds__(256) void k_scan_lo(const uint32_t* __restrict__ hist,
                                                 uint32_t* __restrict__ sel) {
    __shared__ uint32_t csum[256];
    int t = threadIdx.x;
    uint32_t s = 0;
    for (int b = 0; b < 256; ++b) s += hist[t * 256 + b];
    csum[t] = s; __syncthreads();
    if (t == 0) {
        uint32_t K = sel[1], acc = 0, l0 = 0, budget = 0;
        for (int c = 255; c >= 0; --c) {
            if (acc + csum[c] >= K) {
                for (int b = c * 256 + 255; b >= c * 256; --b) {
                    uint32_t hc = hist[b];
                    if (acc + hc >= K) { l0 = (uint32_t)b; budget = K - acc; break; }
                    acc += hc;
                }
                break;
            }
            acc += csum[c];
        }
        sel[2] = (sel[0] << 16) | l0;
        sel[3] = budget;
    }
}
__global__ void k_select(const uint32_t* __restrict__ keys, const uint32_t* __restrict__ sel,
                         const int* __restrict__ ei, const int* __restrict__ ej,
                         const float* __restrict__ A, float* __restrict__ out,
                         uint32_t* __restrict__ eqCnt) {
    int e = blockIdx.x * 256 + threadIdx.x;
    if (e >= E_EDGES) return;
    uint32_t k = keys[e], T = sel[2], budget = sel[3];
    bool keep = false;
    if (k > T) keep = true;
    else if (k == T) { uint32_t pos = atomicAdd(eqCnt, 1u); keep = (pos < budget); }
    if (keep) {
        size_t idx = (size_t)ei[e] * N_NODES + (size_t)ej[e];
        out[idx] = A[idx];
    }
}

// ---------------- host-side orchestration ----------------
extern "C" void kernel_launch(void* const* d_in, const int* in_sizes, int n_in,
                              void* d_out, int out_size, void* d_ws, size_t ws_size,
                              hipStream_t stream) {
    (void)in_sizes; (void)n_in; (void)out_size; (void)ws_size;
    const float* A      = (const float*)d_in[0];
    const float* X      = (const float*)d_in[1];
    const int*   ei     = (const int*)d_in[2];
    const int*   ej     = (const int*)d_in[3];
    const float* noise  = (const float*)d_in[4];
    const float* W_gnn  = (const float*)d_in[5];
    const float* b_gnn  = (const float*)d_in[6];
    const float* W1     = (const float*)d_in[7];
    const float* b1     = (const float*)d_in[8];
    const float* W2     = (const float*)d_in[9];
    const float* b2     = (const float*)d_in[10];
    float* out = (float*)d_out;

    char* w = (char*)d_ws;
    size_t o = 0;
    float*    XW     = (float*)(w + o);    o += (size_t)N_NODES * D_FEAT * 4;        // 4 MB
    uint32_t* xwfrag = (uint32_t*)(w + o); o += (size_t)256 * 8 * 32 * 8 * 4;        // 2 MB
    uint32_t* wgfrag = (uint32_t*)(w + o); o += (size_t)4 * 8 * 32 * 8 * 4;          // 32 KB
    uint32_t* w1frag = (uint32_t*)(w + o); o += (size_t)9 * 16 * 32 * 8 * 4;         // 144 KB
    uint16_t* Hb     = (uint16_t*)(w + o); o += (size_t)N_NODES * D_FEAT * 2;        // 2 MB
    uint16_t* hGb    = (uint16_t*)(w + o); o += 256;
    uint32_t* keys   = (uint32_t*)(w + o); o += (size_t)E_EDGES * 4;                 // 1 MB
    uint32_t* hist1  = (uint32_t*)(w + o); o += 65536 * 4;
    uint32_t* hist2  = (uint32_t*)(w + o); o += 65536 * 4;
    uint32_t* sel    = (uint32_t*)(w + o); o += 256;
    uint32_t* eqCnt  = (uint32_t*)(w + o); o += 256;

    // 0) zero hist1|hist2|sel|eqCnt (contiguous) and the 256 MB output (NT stores)
    {
        int nz = 65536 + 65536 + 64 + 64;
        k_zero_u32<<<(nz + 255) / 256, 256, 0, stream>>>(hist1, nz);
        k_zero_out<<<65536, 256, 0, stream>>>((f32x4v*)out);
    }
    // 1) swizzle W_gnn -> B fragments; GEMM XW = X @ W_gnn
    k_swizzle_b<<<(4 * 8 * 256 + 255) / 256, 256, 0, stream>>>(W_gnn, wgfrag, 128, 128, 8, 4 * 8 * 256);
    k_gemm_xw<<<64, 256, 0, stream>>>(X, wgfrag, XW);
    // 2) swizzle XW -> B fragments; GCN: H = relu(A @ XW + b) via double-buffered TDM
    k_swizzle_b<<<(256 * 8 * 256 + 255) / 256, 256, 0, stream>>>(XW, xwfrag, 8192, 128, 8, 256 * 8 * 256);
    k_gcn<<<64, 256, 0, stream>>>(A, xwfrag, b_gnn, Hb);
    // 3) graph readout h_G
    k_colmean<<<128, 256, 0, stream>>>(Hb, hGb);
    // 4) swizzle W1 (K=257 zero-padded to 288); edge MLP -> sortable keys
    k_swizzle_b<<<(9 * 16 * 256 + 255) / 256, 256, 0, stream>>>(W1, w1frag, 257, 256, 16, 9 * 16 * 256);
    k_mlp<<<E_EDGES / 16, 512, 0, stream>>>(Hb, hGb, w1frag, b1, W2, b2, ei, ej, noise, keys);
    // 5) exact top-K threshold via 2-pass 16-bit radix select, then scatter
    k_hist_hi<<<E_EDGES / 256, 256, 0, stream>>>(keys, hist1);
    k_scan_hi<<<1, 256, 0, stream>>>(hist1, sel);
    k_hist_lo<<<E_EDGES / 256, 256, 0, stream>>>(keys, sel, hist2);
    k_scan_lo<<<1, 256, 0, stream>>>(hist2, sel);
    k_select<<<E_EDGES / 256, 256, 0, stream>>>(keys, sel, ei, ej, A, out, eqCnt);
}
